// StyleBlock_33947421508315
// MI455X (gfx1250) — matrix operand order repaired and checked by reading, not compiled, not run on hardware
//
#include <hip/hip_runtime.h>
#include <hip/hip_bf16.h>
#include <stdint.h>

// ---------------- problem constants ----------------
#define BN   8
#define HH_  128
#define WW_  128
#define CIN  256
#define COUT 256
#define LAT  512

// ---------------- async-to-LDS availability ----------------
#if defined(__has_builtin)
#if __has_builtin(__builtin_amdgcn_global_load_async_to_lds_b128) && \
    __has_builtin(__builtin_amdgcn_s_wait_asynccnt)
#define ATHENA_ASYNC 1
#endif
#endif
#ifndef ATHENA_ASYNC
#define ATHENA_ASYNC 0
#endif

// ---------------- types for WMMA ----------------
typedef __bf16 bf16_t;
typedef bf16_t v16bf __attribute__((ext_vector_type(16)));
typedef float  v8f   __attribute__((ext_vector_type(8)));
typedef unsigned int v4u __attribute__((ext_vector_type(4)));

#if ATHENA_ASYNC
// Exact parameter types per hipcc diagnostic: vector_size(16) int, AS1 src / AS3 dst.
typedef int v4i_vs __attribute__((vector_size(16)));
typedef __attribute__((address_space(1))) v4i_vs* gv4i_p;
typedef __attribute__((address_space(3))) v4i_vs* lv4i_p;
#endif

union ABreg { v16bf v; v4u q[2]; };

// ---------------- bf16 helpers (RNE) ----------------
__device__ __forceinline__ unsigned short f2bf(float f) {
    unsigned u = __float_as_uint(f);
    unsigned r = u + 0x7FFFu + ((u >> 16) & 1u);
    return (unsigned short)(r >> 16);
}
__device__ __forceinline__ unsigned pack2(float lo, float hi) {
    return (unsigned)f2bf(lo) | ((unsigned)f2bf(hi) << 16);
}
__device__ __forceinline__ float lrelu(float x) { return x > 0.f ? x : 0.2f * x; }

// A operand: 16x32 bf16 tile, this code's contiguous k order (k = 16*(lane/16) + s).
// lanes 0-15 : M=lane,   v0-7 -> K 0..15
// lanes 16-31: M=lane-16, v0-7 -> K 16..31
// u32 (bf16-pair) columns: two contiguous groups of 4 => two ds_load_b128.
__device__ __forceinline__ v16bf load_A(const uint32_t* __restrict__ base,
                                        int strideU32, int lane) {
    ABreg a;
    const uint32_t* r = base + (lane & 15) * strideU32 + ((lane >> 4) << 3);
    a.q[0] = *(const v4u*)(r);
    a.q[1] = *(const v4u*)(r + 4);
    return a.v;
}
// B operand: pre-packed [lane][8 x u32] -> two global_load_b128 per lane.
__device__ __forceinline__ v16bf load_B(const uint32_t* __restrict__ p, int lane) {
    ABreg b;
    const v4u* q = (const v4u*)(p + lane * 8);
    b.q[0] = q[0];
    b.q[1] = q[1];
    return b.v;
}

// ================= prep kernels =================

// s[b,i] = latent[b,:] @ stdW[:,i] * sqrt(2/512) + stdb[i]   (2048 threads)
__global__ void k_style(const float* __restrict__ latent, const float* __restrict__ stdW,
                        const float* __restrict__ stdb, float* __restrict__ s) {
    int t = blockIdx.x * 256 + threadIdx.x;
    if (t >= BN * CIN) return;
    int b = t >> 8, i = t & 255;
    float acc = 0.f;
    for (int l = 0; l < LAT; ++l) acc += latent[b * LAT + l] * stdW[l * CIN + i];
    s[t] = acc * 0.0625f + stdb[i];
}

// w2[i,o] = sum_k (w[k,i,o])^2 * (2/2304)   (65536 threads)
__global__ void k_w2(const float* __restrict__ conv_w, float* __restrict__ w2) {
    int t = blockIdx.x * 256 + threadIdx.x;
    if (t >= CIN * COUT) return;
    int i = t >> 8, o = t & 255;
    float acc = 0.f;
#pragma unroll
    for (int kk = 0; kk < 9; ++kk) {
        float w = conv_w[kk * (CIN * COUT) + i * COUT + o];
        acc += w * w;
    }
    w2[t] = acc * (2.0f / 2304.0f);
}

// d[b,o] = rsqrt(sum_i w2[i,o]*s[b,i]^2 + 1e-8)   (2048 threads)
__global__ void k_demod(const float* __restrict__ w2, const float* __restrict__ s,
                        float* __restrict__ dmod) {
    int t = blockIdx.x * 256 + threadIdx.x;
    if (t >= BN * COUT) return;
    int b = t >> 8, o = t & 255;
    float acc = 0.f;
    for (int i = 0; i < CIN; ++i) {
        float sv = s[(b << 8) + i];
        acc += w2[i * COUT + o] * sv * sv;
    }
    dmod[t] = rsqrtf(acc + 1e-8f);
}

// Pack conv weights into WMMA-B lane layout: wp[kk][kc][nt][lane][v]
// B 32x16: lanes<16 K=2v,2v+1 ; lanes>=16 K=16+2v,16+2v+1 ; N=lane&15
__global__ void k_wpack(const float* __restrict__ conv_w, uint32_t* __restrict__ wp) {
    int t = blockIdx.x * 256 + threadIdx.x;       // 294912
    if (t >= 9 * 8 * 16 * 32 * 8) return;
    int v = t & 7, lane = (t >> 3) & 31, nt = (t >> 8) & 15, kc = (t >> 12) & 7, kk = t >> 15;
    int o  = nt * 16 + (lane & 15);
    int k0 = kc * 32 + ((lane & 16) ? 16 : 0) + 2 * v;
    float sc = sqrtf(2.0f / 2304.0f);
    float a = conv_w[kk * (CIN * COUT) + k0 * COUT + o] * sc;
    float b = conv_w[kk * (CIN * COUT) + (k0 + 1) * COUT + o] * sc;
    wp[t] = pack2(a, b);
}

// Pack D1 (256->1024): d1p[n(64)][kc(8)][lane][v]
__global__ void k_d1pack(const float* __restrict__ D1W, uint32_t* __restrict__ d1p) {
    int t = blockIdx.x * 256 + threadIdx.x;       // 131072
    if (t >= 64 * 8 * 32 * 8) return;
    int v = t & 7, lane = (t >> 3) & 31, kc = (t >> 8) & 7, n = t >> 11;
    int o  = n * 16 + (lane & 15);
    int k0 = kc * 32 + ((lane & 16) ? 16 : 0) + 2 * v;
    float sc = 0.08838834764f;  // sqrt(2/256)
    d1p[t] = pack2(D1W[k0 * 1024 + o] * sc, D1W[(k0 + 1) * 1024 + o] * sc);
}

// Pack D2 (1024->256): d2p[n(16)][kc(32)][lane][v]
__global__ void k_d2pack(const float* __restrict__ D2W, uint32_t* __restrict__ d2p) {
    int t = blockIdx.x * 256 + threadIdx.x;       // 131072
    if (t >= 16 * 32 * 32 * 8) return;
    int v = t & 7, lane = (t >> 3) & 31, kc = (t >> 8) & 31, n = t >> 13;
    int o  = n * 16 + (lane & 15);
    int k0 = kc * 32 + ((lane & 16) ? 16 : 0) + 2 * v;
    float sc = 0.04419417382f;  // sqrt(2/1024)
    d2p[t] = pack2(D2W[k0 * COUT + o] * sc, D2W[(k0 + 1) * COUT + o] * sc);
}

// ================= fused modulate + conv + demod + MLP =================
// grid = B*H*(W/32) = 4096 blocks, 256 threads = 8 wave32s.
// LDS: xs[3][34][256] bf16 (52224B) | ys[32][256] bf16 (16384B) | hs[32][1024] bf16 (65536B)
// noise f32 tile (32KB) aliases the first half of hs (consumed before hs is written).
__global__ __launch_bounds__(256) void k_fused(
    const float* __restrict__ data, const float* __restrict__ s,
    const uint32_t* __restrict__ wp,
    const uint32_t* __restrict__ d1p, const uint32_t* __restrict__ d2p,
    const float* __restrict__ dmod, const float* __restrict__ bias,
    const float* __restrict__ ncoef, const float* __restrict__ noise,
    const float* __restrict__ d1b, const float* __restrict__ d2b,
    float* __restrict__ out) {
    extern __shared__ char smem[];
    unsigned short* xs = (unsigned short*)smem;
    unsigned short* ys = (unsigned short*)(smem + 52224);
    unsigned short* hs = (unsigned short*)(smem + 52224 + 16384);
    uint32_t* xs32 = (uint32_t*)xs;
    uint32_t* ys32 = (uint32_t*)ys;
    uint32_t* hs32 = (uint32_t*)hs;
    float* nls = (float*)hs;  // alias: 32x256 f32 noise tile (32KB of 64KB hs)

    int blk = blockIdx.x;
    int wt = blk & 3, hh = (blk >> 2) & 127, b = blk >> 9;
    int w0 = wt * 32;
    int tid = threadIdx.x, lane = tid & 31, wv = tid >> 5;

#if ATHENA_ASYNC
    // ---- async DMA of the contiguous 32KB noise tile into LDS (ASYNCcnt) ----
    {
        const char* nsrc =
            (const char*)(noise + (size_t)((((b << 7) + hh) << 7) + w0) * 256);
        char* ndst = (char*)nls;
        for (int i = tid; i < 2048; i += 256) {  // 2048 x 16B
            __builtin_amdgcn_global_load_async_to_lds_b128(
                (gv4i_p)(uintptr_t)(nsrc + i * 16),
                (lv4i_p)(uintptr_t)(ndst + i * 16), 0, 0);
        }
    }
#endif

    // ---- stage 0: halo patch (3 rows x 34 cols x 256ch), modulate + bf16 ----
    // cp = i & 127 is loop-invariant per thread -> style scalars loaded once.
    const float2* dg = (const float2*)data;
    int cp = tid & 127;
    float sv0 = s[(b << 8) + 2 * cp];
    float sv1 = s[(b << 8) + 2 * cp + 1];
    for (int i = tid; i < 3 * 34 * 128; i += 256) {
        int row = i / (34 * 128);
        int rem = i - row * (34 * 128);
        int col = rem >> 7;
        int h2 = hh + row - 1, w2 = w0 + col - 1;
        uint32_t val = 0;
        if ((unsigned)h2 < 128u && (unsigned)w2 < 128u) {
            float2 xv = dg[(size_t)(((b << 7) + h2) * 128 + w2) * 128 + cp];
            val = pack2(xv.x * sv0, xv.y * sv1);
        }
        xs32[i] = val;
    }
#if ATHENA_ASYNC
    __builtin_amdgcn_s_wait_asynccnt(0);
#endif
    __syncthreads();

    // ---- stage 1: 3x3 conv as implicit GEMM (K = 9*256), 4 tiles per wave ----
    for (int j = 0; j < 4; ++j) {
        int t = wv * 4 + j;
        int mt = t >> 4, nt = t & 15;
        v8f acc = {0.f, 0.f, 0.f, 0.f, 0.f, 0.f, 0.f, 0.f};
        for (int kk = 0; kk < 9; ++kk) {
            int kh = kk / 3, kw = kk - kh * 3;
            const uint32_t* abase = xs32 + (kh * 34 + mt * 16 + kw) * 128;
            const uint32_t* bbase = wp + kk * 32768 + nt * 256;
            __builtin_prefetch(bbase + 32768, 0, 3);  // next-tap weight tile
#pragma unroll
            for (int kc = 0; kc < 8; ++kc) {
                v16bf a  = load_A(abase + kc * 16, 128, lane);
                v16bf bm = load_B(bbase + kc * 4096, lane);
                acc = __builtin_amdgcn_wmma_f32_16x16x32_bf16(
                    false, a, false, bm, (short)0, acc, false, false);
            }
        }
        // epilogue: demod + bias + noise + lrelu -> ys (bf16)
        int o = nt * 16 + (lane & 15);
        int hm = (lane >> 4) << 3;
        float dv = dmod[(b << 8) + o];
        float bs = bias[o];
        float nc = ncoef[o];
#pragma unroll
        for (int r = 0; r < 8; ++r) {
            int m  = mt * 16 + r + hm;
#if ATHENA_ASYNC
            float nz = nls[m * 256 + o];
#else
            int ww = w0 + m;
            float nz = noise[((((b << 7) + hh) << 7) + ww) * 256 + o];
#endif
            ys[m * 256 + o] = f2bf(lrelu(acc[r] * dv + bs + nz * nc));
        }
    }
    __syncthreads();

    // ---- stage 2: h = lrelu(y @ D1 + b1), 8 N-tiles x 2 M-tiles per wave ----
    for (int j = 0; j < 8; ++j) {
        int n2 = wv * 8 + j;
        const uint32_t* bb = d1p + n2 * 2048;  // (n*8+kc)*256
        int o1 = n2 * 16 + (lane & 15);
        float bsv = d1b[o1];
        for (int mt = 0; mt < 2; ++mt) {
            v8f acc = {0.f, 0.f, 0.f, 0.f, 0.f, 0.f, 0.f, 0.f};
#pragma unroll
            for (int kc = 0; kc < 8; ++kc) {
                v16bf a  = load_A(ys32 + (mt * 16) * 128 + kc * 16, 128, lane);
                v16bf bm = load_B(bb + kc * 256, lane);
                acc = __builtin_amdgcn_wmma_f32_16x16x32_bf16(
                    false, a, false, bm, (short)0, acc, false, false);
            }
            int hm = (lane >> 4) << 3;
#pragma unroll
            for (int r = 0; r < 8; ++r) {
                int m = mt * 16 + r + hm;
                hs[m * 1024 + o1] = f2bf(lrelu(acc[r] + bsv));
            }
        }
    }
    __syncthreads();

    // ---- stage 3: out = lrelu(h @ D2 + b2), 4 tiles per wave, K=1024 ----
    for (int j = 0; j < 4; ++j) {
        int t = wv * 4 + j;
        int mt = t >> 4, nt = t & 15;
        const uint32_t* bb = d2p + nt * 8192;  // (n*32+kc)*256
        v8f acc = {0.f, 0.f, 0.f, 0.f, 0.f, 0.f, 0.f, 0.f};
#pragma unroll 8
        for (int kc = 0; kc < 32; ++kc) {
            v16bf a  = load_A(hs32 + (mt * 16) * 512 + kc * 16, 512, lane);
            v16bf bm = load_B(bb + kc * 256, lane);
            acc = __builtin_amdgcn_wmma_f32_16x16x32_bf16(
                false, a, false, bm, (short)0, acc, false, false);
        }
        int o = nt * 16 + (lane & 15);
        float bsv = d2b[o];
        int hm = (lane >> 4) << 3;
#pragma unroll
        for (int r = 0; r < 8; ++r) {
            int m  = mt * 16 + r + hm;
            int ww = w0 + m;
            out[((((b << 7) + hh) << 7) + ww) * 256 + o] = lrelu(acc[r] + bsv);
        }
    }
}

// ================= launch =================
extern "C" void kernel_launch(void* const* d_in, const int* in_sizes, int n_in,
                              void* d_out, int out_size, void* d_ws, size_t ws_size,
                              hipStream_t stream) {
    const float* data   = (const float*)d_in[0];
    const float* latent = (const float*)d_in[1];
    const float* noise  = (const float*)d_in[2];
    const float* stdW   = (const float*)d_in[3];
    const float* stdb   = (const float*)d_in[4];
    const float* conv_w = (const float*)d_in[5];
    const float* bias   = (const float*)d_in[6];
    const float* ncoef  = (const float*)d_in[7];
    const float* D1W    = (const float*)d_in[8];
    const float* D1b    = (const float*)d_in[9];
    const float* D2W    = (const float*)d_in[10];
    const float* D2b    = (const float*)d_in[11];
    float* out = (float*)d_out;

    char* ws = (char*)d_ws;
    uint32_t* wp   = (uint32_t*)ws; ws += 9 * 8 * 16 * 32 * 8 * 4;   // 1.12 MB
    uint32_t* d1p  = (uint32_t*)ws; ws += 64 * 8 * 32 * 8 * 4;       // 512 KB
    uint32_t* d2p  = (uint32_t*)ws; ws += 16 * 32 * 32 * 8 * 4;      // 512 KB
    float* s    = (float*)ws; ws += BN * CIN * 4;
    float* w2   = (float*)ws; ws += CIN * COUT * 4;
    float* dmod = (float*)ws; ws += BN * COUT * 4;

    k_style<<<8, 256, 0, stream>>>(latent, stdW, stdb, s);
    k_w2<<<256, 256, 0, stream>>>(conv_w, w2);
    k_demod<<<8, 256, 0, stream>>>(w2, s, dmod);
    k_wpack<<<1152, 256, 0, stream>>>(conv_w, wp);
    k_d1pack<<<512, 256, 0, stream>>>(D1W, d1p);
    k_d2pack<<<512, 256, 0, stream>>>(D2W, d2p);

    int lds_bytes = 52224 + 16384 + 65536;  // 131 KB < 320 KB/WGP
    k_fused<<<BN * HH_ * (WW_ / 32), 256, lds_bytes, stream>>>(
        data, s, wp, d1p, d2p, dmod, bias, ncoef, noise, D1b, D2b, out);
}
